// Conv2Demod_23467701305348
// MI455X (gfx1250) — compile-verified, matrix operand order checked by reading, not run on hardware
//
#include <hip/hip_runtime.h>

// ---------------------------------------------------------------------------
// StyleGAN2 modulated 3x3 conv for MI455X (gfx1250, wave32, WMMA).
// y[b,o,h,w] = dcoef[b,o] * conv3x3(img[b]*styles[b], weight)[o,h,w]
//            + B_param[o]*noise[b,h,w]
// Heavy conv: implicit GEMM on v_wmma_f32_16x16x32_bf16, with double-buffered
// GLOBAL_LOAD_ASYNC_TO_LDS staging (ASYNCcnt) to overlap fetch with WMMA.
// ---------------------------------------------------------------------------

#define B_SZ 16
#define C_IN 512
#define C_OUT 512
#define Z_DIM 512
#define KK 3
#define HH 64
#define WW 64
#define HW (HH * WW)

#define ABUF_ELEMS (9 * 64 * 32)   // [tap][co][ci]   bf16
#define XBUF_ELEMS (3 * 66 * 32)   // [dh][col+1][ci] bf16

typedef __attribute__((ext_vector_type(16))) __bf16 v16bf;
typedef __attribute__((ext_vector_type(8)))  float  v8f;

union U16 { uint4 u[2]; v16bf v; };

__device__ __forceinline__ unsigned short f2bf(float f) {
    union { float f; unsigned u; } x;
    x.f = f;
    unsigned u = x.u;
    unsigned r = u + 0x7FFFu + ((u >> 16) & 1u);   // round-to-nearest-even
    return (unsigned short)(r >> 16);
}

// ---- styles[b][i] = ws[b] . A_w[i] + A_b[i] -------------------------------
__global__ void k_styles(const float* __restrict__ ws, const float* __restrict__ Aw,
                         const float* __restrict__ Ab, float* __restrict__ styles) {
    int t = blockIdx.x * blockDim.x + threadIdx.x;      // 0 .. B_SZ*C_IN-1
    int b = t >> 9, i = t & 511;
    const float* wr = ws + b * Z_DIM;
    const float* ar = Aw + i * Z_DIM;
    float acc = Ab[i];
    for (int k = 0; k < Z_DIM; ++k) acc += wr[k] * ar[k];
    styles[t] = acc;
}

// ---- bf16 weight [tap][co][ci] + W2[co][ci] = sum_tap w^2 -----------------
__global__ void k_wprep(const float* __restrict__ wt, unsigned short* __restrict__ wbf,
                        float* __restrict__ W2) {
    int t = blockIdx.x * blockDim.x + threadIdx.x;      // 0 .. C_OUT*C_IN-1
    int co = t >> 9, ci = t & 511;
    const float* p = wt + (size_t)(co * C_IN + ci) * (KK * KK);
    float s = 0.f;
#pragma unroll
    for (int k = 0; k < KK * KK; ++k) {
        float v = p[k];
        s += v * v;
        wbf[((size_t)k * C_OUT + co) * C_IN + ci] = f2bf(v);
    }
    W2[t] = s;
}

// ---- dcoef[b][o] = rsqrt(sum_i W2[o][i]*styles[b][i]^2 + eps) -------------
__global__ void k_dcoef(const float* __restrict__ W2, const float* __restrict__ styles,
                        float* __restrict__ dcoef) {
    int t = blockIdx.x * blockDim.x + threadIdx.x;      // 0 .. B_SZ*C_OUT-1
    int b = t >> 9, o = t & 511;
    const float* w2 = W2 + o * C_IN;
    const float* st = styles + b * C_IN;
    float acc = 0.f;
    for (int i = 0; i < C_IN; ++i) { float s = st[i]; acc += w2[i] * s * s; }
    dcoef[t] = rsqrtf(acc + 1e-8f);
}

// ---- modulated input -> bf16 NHWC: xbf[b][h][w][ci] -----------------------
__global__ void k_xmod(const float* __restrict__ img, const float* __restrict__ styles,
                       unsigned* __restrict__ xpk) {
    int tid = blockIdx.x * blockDim.x + threadIdx.x;    // pair index, NHWC order
    int cp = tid & 255;                                 // ci pair
    int r  = tid >> 8;
    int w  = r & 63; r >>= 6;
    int h  = r & 63; int b = r >> 6;
    int ci = cp << 1;
    float s0 = styles[b * C_IN + ci], s1 = styles[b * C_IN + ci + 1];
    const float* ip = img + (((size_t)(b * C_IN + ci)) * HH + h) * WW + w;
    unsigned short l0 = f2bf(ip[0] * s0);
    unsigned short l1 = f2bf(ip[HW] * s1);
    xpk[tid] = (unsigned)l0 | ((unsigned)l1 << 16);
}

// ---- async staging of one K-chunk into LDS buffer `buf` -------------------
__device__ __forceinline__ void issue_chunk(
        const unsigned short* __restrict__ wbf, const unsigned short* __restrict__ xbf,
        unsigned abase, unsigned xbase,          // LDS byte offsets of A/X buffers
        int co_base, int b, int h, int kc, int tid) {
    // weights: 9 taps x 64 co x 32 ci  => 2304 x 16B lane-tasks
#pragma unroll
    for (int j = 0; j < 9; ++j) {
        int id = j * 256 + tid;
        int r = id >> 2, q = id & 3;
        int tap = r >> 6, co = r & 63;
        const unsigned short* g = &wbf[(((size_t)tap * C_OUT + co_base + co) * C_IN)
                                       + kc * 32 + q * 8];
        unsigned loff = abase + (unsigned)(((tap * 64 + co) * 32 + q * 8) * 2);
        asm volatile("global_load_async_to_lds_b128 %0, %1, off"
                     :: "v"(loff), "v"(g) : "memory");
    }
    // input slab: rows h-1..h+1, image cols 0..63 -> LDS cols 1..64
#pragma unroll
    for (int j = 0; j < 3; ++j) {                // j == dh (wave-uniform)
        int id = j * 256 + tid;
        int r = id >> 2, q = id & 3;
        int col = (r & 63) + 1;
        int hh = h + j - 1;
        if (hh >= 0 && hh < HH) {                // uniform per block/j
            const unsigned short* g = &xbf[(((size_t)(b * HH + hh)) * WW + (col - 1)) * C_IN
                                           + kc * 32 + q * 8];
            unsigned loff = xbase + (unsigned)(((j * 66 + col) * 32 + q * 8) * 2);
            asm volatile("global_load_async_to_lds_b128 %0, %1, off"
                         :: "v"(loff), "v"(g) : "memory");
        }
    }
}

// ---- main conv: per block = (batch b, row h, 64-wide C_OUT tile) ----------
// C[64co x 64w] = sum_{tap,ci-chunk} Wt[64x32] @ Xshift[32x64]  (bf16 WMMA)
__global__ __launch_bounds__(256) void k_conv(
        const unsigned short* __restrict__ wbf, const unsigned short* __restrict__ xbf,
        const float* __restrict__ dcoef, const float* __restrict__ Bp,
        const float* __restrict__ noise, float* __restrict__ out) {
    extern __shared__ __align__(16) unsigned short smem[];
    // layout (ushort elems): A0 [0,18432) A1 [18432,36864) X0 [36864,43200) X1 [43200,49536)

    const int bid = blockIdx.x;
    const int mtile = bid & 7;
    const int h = (bid >> 3) & 63;
    const int b = bid >> 9;
    const int co_base = mtile * 64;

    const int tid  = threadIdx.x;
    const int wave = tid >> 5;
    const int lane = tid & 31;
    const int m    = wave >> 1;                 // M tile 0..3
    const int nb0  = (wave & 1) * 32;           // two N tiles per wave
    const int nb1  = nb0 + 16;
    const int row  = lane & 15;
    const int k0   = (lane & 16) ? 8 : 0;       // 16-bit fragment K interleave

    const unsigned smem_base = (unsigned)(uintptr_t)smem;   // LDS byte offset
    unsigned short* X0 = smem + 2 * ABUF_ELEMS;

    v8f acc0 = {}, acc1 = {};

    // ---- one-time zero fill of regions async copies never write ----------
    // halo columns (LDS col 0 and 65) of dh slabs 0 and 2 only, in both X buffers
    if (tid < 48) {
        int bufi = tid >= 24, t = tid & 23;
        int dh = t >> 3, wi = t & 7;
        int c  = (wi >= 4) ? 65 : 0, q = wi & 3;
        uint4 z = {0u, 0u, 0u, 0u};
        *(uint4*)&X0[bufi * XBUF_ELEMS + (dh * 66 + c) * 32 + q * 8] = z;
    }
    // out-of-range image rows (h==0 -> dh slab 0, h==63 -> dh slab 2)
    if (h == 0 || h == 63) {
        int dh = (h == 0) ? 0 : 2;
        uint4 z = {0u, 0u, 0u, 0u};
        for (int idx = tid; idx < 2 * 264; idx += 256) {   // 264 uint4 per slab
            int bufi = idx >= 264, e = (idx & 255) + (idx >= 264 ? idx - 264 - (idx & 255) + (idx & 255) : 0);
            e = idx % 264;
            *(uint4*)&X0[bufi * XBUF_ELEMS + dh * 66 * 32 + e * 8] = z;
        }
    }

    // ---- prologue: fetch chunk 0 into buffer 0 ---------------------------
    issue_chunk(wbf, xbf, smem_base, smem_base + 2 * ABUF_ELEMS * 2, co_base, b, h, 0, tid);

    for (int kc = 0; kc < C_IN / 32; ++kc) {
        const int cur = kc & 1;
        asm volatile("s_wait_asynccnt 0x0" ::: "memory");   // my current-buffer loads done
        __syncthreads();                                    // everyone's loads + prev reads done
        if (kc + 1 < C_IN / 32) {
            const int nxt = cur ^ 1;
            issue_chunk(wbf, xbf,
                        smem_base + (unsigned)(nxt * ABUF_ELEMS * 2),
                        smem_base + (unsigned)((2 * ABUF_ELEMS + nxt * XBUF_ELEMS) * 2),
                        co_base, b, h, kc + 1, tid);
        }

        const unsigned short* As = smem + cur * ABUF_ELEMS;
        const unsigned short* Xs = smem + 2 * ABUF_ELEMS + cur * XBUF_ELEMS;

#pragma unroll
        for (int dh = 0; dh < 3; ++dh) {
#pragma unroll
            for (int dw = 0; dw < 3; ++dw) {
                const int t = dh * 3 + dw;
                U16 au, b0u, b1u;
                const unsigned short* ap = &As[(t * 64 + m * 16 + row) * 32 + k0];
                au.u[0] = *(const uint4*)ap;
                au.u[1] = *(const uint4*)(ap + 16);
                const unsigned short* bp0 = &Xs[((dh * 66) + nb0 + row + dw) * 32 + k0];
                b0u.u[0] = *(const uint4*)bp0;
                b0u.u[1] = *(const uint4*)(bp0 + 16);
                const unsigned short* bp1 = &Xs[((dh * 66) + nb1 + row + dw) * 32 + k0];
                b1u.u[0] = *(const uint4*)bp1;
                b1u.u[1] = *(const uint4*)(bp1 + 16);
                acc0 = __builtin_amdgcn_wmma_f32_16x16x32_bf16(
                           false, au.v, false, b0u.v, (short)0, acc0, false, false);
                acc1 = __builtin_amdgcn_wmma_f32_16x16x32_bf16(
                           false, au.v, false, b1u.v, (short)0, acc1, false, false);
            }
        }
    }

    // epilogue: demodulate + noise, C layout: VGPR j -> M=j (lanes 0-15) / j+8
    const int col0 = nb0 + row, col1 = nb1 + row;
    const float n0v = noise[(b * HH + h) * WW + col0];
    const float n1v = noise[(b * HH + h) * WW + col1];
    const int mbase = co_base + m * 16 + ((lane & 16) ? 8 : 0);
#pragma unroll
    for (int j = 0; j < 8; ++j) {
        int co = mbase + j;
        float d  = dcoef[b * C_OUT + co];
        float bp = Bp[co];
        size_t o = (((size_t)(b * C_OUT + co)) * HH + h) * WW;
        out[o + col0] = acc0[j] * d + bp * n0v;
        out[o + col1] = acc1[j] * d + bp * n1v;
    }
}

extern "C" void kernel_launch(void* const* d_in, const int* in_sizes, int n_in,
                              void* d_out, int out_size, void* d_ws, size_t ws_size,
                              hipStream_t stream) {
    const float* img   = (const float*)d_in[0];
    const float* ws    = (const float*)d_in[1];
    const float* noise = (const float*)d_in[2];
    const float* wt    = (const float*)d_in[3];
    const float* Aw    = (const float*)d_in[4];
    const float* Ab    = (const float*)d_in[5];
    const float* Bp    = (const float*)d_in[6];
    float* out = (float*)d_out;

    char* wsb = (char*)d_ws;
    float*          styles = (float*)(wsb + 0);                        // 32 KB
    float*          dcoef  = (float*)(wsb + 32768);                    // 32 KB
    float*          W2     = (float*)(wsb + 65536);                    // 1 MB
    unsigned short* wbf    = (unsigned short*)(wsb + 1114112);         // 4.5 MB
    unsigned short* xbf    = (unsigned short*)(wsb + 5832704);         // 64 MB

    k_styles<<<(B_SZ * C_IN) / 256, 256, 0, stream>>>(ws, Aw, Ab, styles);
    k_wprep <<<(C_OUT * C_IN) / 256, 256, 0, stream>>>(wt, wbf, W2);
    k_dcoef <<<(B_SZ * C_OUT) / 256, 256, 0, stream>>>(W2, styles, dcoef);
    k_xmod  <<<(B_SZ * HH * WW * C_IN / 2) / 256, 256, 0, stream>>>(img, styles,
                                                                    (unsigned*)xbf);
    const size_t lds_bytes = (2 * ABUF_ELEMS + 2 * XBUF_ELEMS) * sizeof(unsigned short);
    k_conv  <<<B_SZ * HH * (C_OUT / 64), 256, lds_bytes, stream>>>(wbf, xbf, dcoef, Bp,
                                                                   noise, out);
}
